// SimpleRNN9_22660247454200
// MI455X (gfx1250) — matrix operand order repaired and checked by reading, not compiled, not run on hardware
//
#include <hip/hip_runtime.h>

typedef __attribute__((ext_vector_type(16))) _Float16 v16h;
typedef __attribute__((ext_vector_type(8)))  float    v8f;

#define NB    4096
#define NK    512
#define NU    4
#define NOBS_ 3
#define NLIFT 32
#define NHID  128
#define NP    9
#define BT    16   // batch rows per block

#define WMMA_F32(a,b,c) __builtin_amdgcn_wmma_f32_16x16x32_f16(false,(a),false,(b),(short)0,(c),false,false)

// fast transcendentals: v_rcp_f32 / v_exp_f32 (TRANS ops), no IEEE divide expansion
__device__ __forceinline__ float frcp(float x){ return __builtin_amdgcn_rcpf(x); }
__device__ __forceinline__ float sigf(float x){ return frcp(1.0f + __expf(-x)); }
__device__ __forceinline__ float tanh_fast(float x){ return 1.0f - 2.0f*frcp(1.0f + __expf(2.0f*x)); }

// A-matrix (16xK f16) fragment from row-major LDS, contiguous K order matching B:
// VGPR j (0..7): lanes0-15 K=2j,2j+1 ; lanes16-31 K=16+2j,16+2j+1
__device__ __forceinline__ v16h afrag_lds(const _Float16* A, int stride, int m, int kbase, bool lo){
  v16h a;
  #pragma unroll
  for (int j=0;j<8;++j){
    int kk = kbase + (lo ? 0 : 16) + 2*j;
    a[2*j]   = A[m*stride+kk];
    a[2*j+1] = A[m*stride+kk+1];
  }
  return a;
}

// B-matrix (32x16) fragment: lane = column N; lanes0-15 hold K=kbase..kbase+15,
// lanes16-31 hold K=kbase+16..kbase+31.  Source W is row-major [N][K] (W^T feed).
__device__ __forceinline__ v16h bfrag_global(const float* W, int K, int nbase, int kbase, bool lo, int ln){
  v16h b;
  const float* row = W + (size_t)(nbase+ln)*K + kbase + (lo?0:16);
  #pragma unroll
  for (int v=0;v<16;++v) b[v] = (_Float16)row[v];
  return b;
}

__device__ __forceinline__ v16h bfrag_lds(const _Float16* W, int K, int nbase, int kbase, bool lo, int ln){
  v16h b;
  const _Float16* row = W + (nbase+ln)*K + kbase + (lo?0:16);
  #pragma unroll
  for (int v=0;v<16;++v) b[v] = row[v];
  return b;
}

__device__ __forceinline__ void rhs9(const float* y, const float* th, float* d){
  float A=y[0],D=y[1],G=y[2],H=y[3],I=y[4],J=y[5],Kv=y[6],L=y[7],M=y[8];
  float kfAD=th[0],kfDG=th[1],kf7=th[2],kf8=th[3],kf9=th[4],kf10=th[5],kf11=th[6],kf12=th[7];
  float krAD=th[8],krDG=th[9],kr7=th[10],kr9=th[11],kr11=th[12],kr12=th[13];
  d[0] = -kfAD*A + krAD*D;
  d[1] =  kfAD*A - krAD*D - kfDG*D + krDG*G;
  d[2] =  kfDG*D - krDG*G - kf7*G + kr7*H;
  d[3] =  kf7*G - kr7*H - kf8*H;
  d[4] =  kf8*H - kf9*I + kr9*J;
  d[5] =  kf9*I - kr9*J - kf10*J;
  d[6] =  kf10*J - kf11*Kv + kr11*L;
  d[7] =  kf11*Kv - kr11*L - kf12*L + kr12*M;
  d[8] =  kf12*L - kr12*M;
}

__device__ __forceinline__ void rk4_step(float* x, float dt, const float* th){
  float k1[NP],k2[NP],k3[NP],k4[NP],t[NP];
  rhs9(x,th,k1);
  #pragma unroll
  for (int j=0;j<NP;++j) t[j]=x[j]+0.5f*dt*k1[j];
  rhs9(t,th,k2);
  #pragma unroll
  for (int j=0;j<NP;++j) t[j]=x[j]+0.5f*dt*k2[j];
  rhs9(t,th,k3);
  #pragma unroll
  for (int j=0;j<NP;++j) t[j]=x[j]+dt*k3[j];
  rhs9(t,th,k4);
  #pragma unroll
  for (int j=0;j<NP;++j)
    x[j]=fmaxf(x[j]+dt*(1.0f/6.0f)*(k1[j]+2.0f*k2[j]+2.0f*k3[j]+k4[j]), 0.0f);
}

__global__ __launch_bounds__(256) void rnn9_scan_kernel(
    const float* __restrict__ y0_obs, const float* __restrict__ u_seq,
    const float* __restrict__ dt_seq, const float* __restrict__ y_seq,
    const float* __restrict__ liftW,  const float* __restrict__ liftb,
    const float* __restrict__ Wih,    const float* __restrict__ Whh,
    const float* __restrict__ bih,    const float* __restrict__ bhh,
    const float* __restrict__ headW,  const float* __restrict__ headb,
    const float* __restrict__ uxj,    const int*   __restrict__ obs,
    float* __restrict__ y_hat, float* __restrict__ theta_hat)
{
  __shared__ _Float16 h16[BT*NHID];     // h state, f16, WMMA A feed
  __shared__ float    hf32[BT*NHID];    // h state, f32 shadow
  __shared__ _Float16 zl16[BT*NLIFT];
  __shared__ _Float16 hW16[16*NHID];    // head_W padded 14->16 rows
  __shared__ float feat_s[BT*8];
  __shared__ float theta_s[BT*16];
  __shared__ float x_s[BT*12];
  __shared__ float br_s[NHID], bz_s[NHID], bin_s[NHID], bhn_s[NHID];
  __shared__ float liftW_s[NLIFT*7], liftb_s[NLIFT];
  __shared__ float uxj_s[NU*NP];
  __shared__ float headb_s[16];
  __shared__ int   obs_s[4];

  const int tid  = threadIdx.x;
  const int lane = tid & 31;
  const int wave = tid >> 5;          // 0..7
  const bool lo  = lane < 16;
  const int ln   = lane & 15;
  const int b0   = blockIdx.x * BT;

  // ---------------- one-time staging ----------------
  for (int i=tid;i<BT*NHID;i+=256){ h16[i]=(_Float16)0.0f; hf32[i]=0.0f; }
  for (int i=tid;i<16*NHID;i+=256){
    int n=i/NHID, kk=i%NHID;
    hW16[i] = (_Float16)((n<14) ? headW[n*NHID+kk] : 0.0f);
  }
  for (int i=tid;i<NHID;i+=256){
    br_s[i]  = bih[i]        + bhh[i];
    bz_s[i]  = bih[NHID+i]   + bhh[NHID+i];
    bin_s[i] = bih[2*NHID+i];
    bhn_s[i] = bhh[2*NHID+i];
  }
  for (int i=tid;i<NLIFT*7;i+=256) liftW_s[i]=liftW[i];
  if (tid<NLIFT)  liftb_s[tid]=liftb[tid];
  if (tid<NU*NP)  uxj_s[tid]=uxj[tid];
  if (tid<16)     headb_s[tid] = (tid<14)?headb[tid]:0.0f;
  if (tid<NOBS_)  obs_s[tid]=obs[tid];
  __syncthreads();
  if (tid<BT){
    int m=tid;
    #pragma unroll
    for (int j=0;j<NP;++j) x_s[m*12+j]=0.01f;
    #pragma unroll
    for (int i=0;i<NOBS_;++i)
      x_s[m*12+obs_s[i]] = y0_obs[(size_t)(b0+m)*NOBS_+i] + 0.01f;
  }

  // Register-resident weight fragments: wave w owns N-tiles {w, 8+w, 16+w}
  // so r/z/n gates for hid columns [16w,16w+16) are all wave-local.
  v16h bWih_r = bfrag_global(Wih, NLIFT,          16*wave, 0, lo, ln);
  v16h bWih_z = bfrag_global(Wih, NLIFT,   NHID + 16*wave, 0, lo, ln);
  v16h bWih_n = bfrag_global(Wih, NLIFT, 2*NHID + 16*wave, 0, lo, ln);
  v16h bWhh_r[4], bWhh_z[4], bWhh_n[4];
  #pragma unroll
  for (int c=0;c<4;++c){
    bWhh_r[c]=bfrag_global(Whh, NHID,          16*wave, 32*c, lo, ln);
    bWhh_z[c]=bfrag_global(Whh, NHID,   NHID + 16*wave, 32*c, lo, ln);
    bWhh_n[c]=bfrag_global(Whh, NHID, 2*NHID + 16*wave, 32*c, lo, ln);
  }
  __syncthreads();

  // ---------------- sequential scan over K ----------------
  for (int k=0;k<NK;++k){
    // phase 1: features (teacher forcing every 50 steps)
    if (tid<BT){
      int m=tid;
      const float* u = u_seq + ((size_t)(b0+m)*NK + k)*NU;
      #pragma unroll
      for (int i=0;i<NU;++i) feat_s[m*8+i]=__builtin_nontemporal_load(u+i);
      bool tf = (k>0) && (k%50==0);
      #pragma unroll
      for (int i=0;i<NOBS_;++i){
        float v = tf ? __builtin_nontemporal_load(
                          y_seq + ((size_t)(b0+m)*NK + (k-1))*NOBS_ + i)
                     : x_s[m*12+obs_s[i]];
        feat_s[m*8+NU+i]=v;
      }
    }
    __syncthreads();
    // lift + SiLU: 16x32 outputs, 2 per thread
    #pragma unroll
    for (int rep=0;rep<2;++rep){
      int idx = tid + rep*256;
      int m = idx >> 5, j = idx & 31;
      float acc = liftb_s[j];
      #pragma unroll
      for (int t=0;t<7;++t) acc += feat_s[m*8+t]*liftW_s[j*7+t];
      zl16[m*NLIFT+j] = (_Float16)(acc*sigf(acc));
    }
    __syncthreads();

    // phase 2: GRU GEMMs via WMMA (f32 accumulate)
    v8f acc_r={},acc_z={},acc_ni={},acc_nh={};
    {
      v16h a = afrag_lds(zl16, NLIFT, ln, 0, lo);
      acc_r  = WMMA_F32(a, bWih_r, acc_r);
      acc_z  = WMMA_F32(a, bWih_z, acc_z);
      acc_ni = WMMA_F32(a, bWih_n, acc_ni);
      #pragma unroll
      for (int c=0;c<4;++c){
        v16h ah = afrag_lds(h16, NHID, ln, 32*c, lo);
        acc_r  = WMMA_F32(ah, bWhh_r[c], acc_r);
        acc_z  = WMMA_F32(ah, bWhh_z[c], acc_z);
        acc_nh = WMMA_F32(ah, bWhh_n[c], acc_nh);
      }
    }
    __syncthreads();  // all reads of h16 complete before rewrite

    // phase 3: gate math (wave-local), rewrite h
    {
      int j = 16*wave + ln;
      #pragma unroll
      for (int v=0;v<8;++v){
        int m = v + (lo?0:8);
        float r  = sigf(acc_r[v] + br_s[j]);
        float z  = sigf(acc_z[v] + bz_s[j]);
        float n  = tanh_fast(acc_ni[v] + bin_s[j] + r*(acc_nh[v]+bhn_s[j]));
        float hp = hf32[m*NHID+j];
        float hn = (1.0f-z)*n + z*hp;
        hf32[m*NHID+j]=hn;
        h16[m*NHID+j]=(_Float16)hn;
      }
    }
    __syncthreads();

    // phase 4: head GEMM (wave 0), theta
    if (wave==0){
      v8f acc={};
      #pragma unroll
      for (int c=0;c<4;++c){
        v16h ah = afrag_lds(h16, NHID, ln, 32*c, lo);
        v16h bh = bfrag_lds(hW16, NHID, 0, 32*c, lo, ln);
        acc = WMMA_F32(ah, bh, acc);
      }
      #pragma unroll
      for (int v=0;v<8;++v){
        int m=v+(lo?0:8);
        if (ln<14){
          float raw = acc[v] + headb_s[ln];
          float th  = 0.01f + 2.99f*sigf(raw);
          theta_s[m*16+ln]=th;
          __builtin_nontemporal_store(
              th, theta_hat + ((size_t)(b0+m)*NK + k)*14 + ln);
        }
      }
    }
    __syncthreads();

    // phase 5: jump + 10x RK4 (one lane per batch row)
    if (tid<BT){
      int m=tid;
      float th[14];
      #pragma unroll
      for (int i=0;i<14;++i) th[i]=theta_s[m*16+i];
      float x[NP];
      #pragma unroll
      for (int j=0;j<NP;++j) x[j]=x_s[m*12+j];
      const float* u = u_seq + ((size_t)(b0+m)*NK + k)*NU;
      #pragma unroll
      for (int j=0;j<NP;++j){
        float s=0.0f;
        #pragma unroll
        for (int i=0;i<NU;++i) s += feat_s[m*8+i]*uxj_s[i*NP+j];
        x[j]+=s;
      }
      (void)u;
      float dt = __builtin_nontemporal_load(dt_seq + (size_t)(b0+m)*NK + k);
      float hsub = dt*0.1f;
      for (int sub=0;sub<10;++sub) rk4_step(x, hsub, th);
      #pragma unroll
      for (int j=0;j<NP;++j) x_s[m*12+j]=x[j];
      #pragma unroll
      for (int i=0;i<NOBS_;++i)
        __builtin_nontemporal_store(
            x[obs_s[i]], y_hat + ((size_t)(b0+m)*NK + k)*NOBS_ + i);
    }
    __syncthreads();
  }
}

extern "C" void kernel_launch(void* const* d_in, const int* in_sizes, int n_in,
                              void* d_out, int out_size, void* d_ws, size_t ws_size,
                              hipStream_t stream) {
  (void)in_sizes; (void)n_in; (void)out_size; (void)d_ws; (void)ws_size;
  const float* y0    = (const float*)d_in[0];
  const float* u_seq = (const float*)d_in[1];
  const float* dtseq = (const float*)d_in[2];
  const float* y_seq = (const float*)d_in[3];
  const float* liftW = (const float*)d_in[4];
  const float* liftb = (const float*)d_in[5];
  const float* Wih   = (const float*)d_in[6];
  const float* Whh   = (const float*)d_in[7];
  const float* bih   = (const float*)d_in[8];
  const float* bhh   = (const float*)d_in[9];
  const float* headW = (const float*)d_in[10];
  const float* headb = (const float*)d_in[11];
  const float* uxj   = (const float*)d_in[12];
  const int*   obs   = (const int*)d_in[13];

  float* out = (float*)d_out;
  float* y_hat     = out;                          // (B,K,3)
  float* theta_hat = out + (size_t)NB*NK*3;        // (B,K,14)

  rnn9_scan_kernel<<<NB/BT, 256, 0, stream>>>(
      y0, u_seq, dtseq, y_seq, liftW, liftb, Wih, Whh, bih, bhh,
      headW, headb, uxj, obs, y_hat, theta_hat);
}
